// FasterMultiHeadAttention_42545946034286
// MI455X (gfx1250) — matrix operand order repaired and checked by reading, not compiled, not run on hardware
//
#include <hip/hip_runtime.h>
#include <hip/hip_bf16.h>

typedef __attribute__((ext_vector_type(16))) _Float16 v16h;
typedef __attribute__((ext_vector_type(8)))  _Float16 v8h;
typedef __attribute__((ext_vector_type(8)))  float    v8f;

#define B_  16
#define N_  1024
#define H_  768
#define NH_ 12
#define HD_ 64

// ---- WMMA fragment helpers (layouts per CDNA5 ISA, wave32) ----
// A 16x32 f16: lane L holds row (L&15); elems 0..7 = K[kh*8 .. kh*8+7],
// elems 8..15 = K[16+kh*8 .. 16+kh*8+7], kh = L>>4.  Row-major [M][K] source.
static __device__ inline v16h load_a_frag_h(const _Float16* base, int ld, int lane) {
  const int r = lane & 15, kh = lane >> 4;
  const _Float16* p = base + (size_t)r * ld + kh * 8;
  v8h lo = *(const v8h*)(p);
  v8h hi = *(const v8h*)(p + 16);
  v16h out;
#pragma unroll
  for (int i = 0; i < 8; ++i) { out[i] = lo[i]; out[i + 8] = hi[i]; }
  return out;
}

static __device__ inline v16h load_a_frag_f32(const float* base, int ld, int lane) {
  const int r = lane & 15, kh = lane >> 4;
  const float* p = base + (size_t)r * ld + kh * 8;
  v16h out;
#pragma unroll
  for (int i = 0; i < 8; ++i) { out[i] = (_Float16)p[i]; out[i + 8] = (_Float16)p[i + 16]; }
  return out;
}

// B 32x16 f16: lane L holds column (L&15); elems 0..7 = K[kh*8 .. kh*8+7],
// elems 8..15 = K[16+kh*8 .. 16+kh*8+7], kh = L>>4.
// Source must be K-transposed storage BT[n][k] (row-major over n, leading dim = ld).
static __device__ inline v16h load_b_frag(const _Float16* baseT, int ld, int lane) {
  const int c = lane & 15, kh = lane >> 4;
  const _Float16* p = baseT + (size_t)c * ld + kh * 8;
  v8h lo = *(const v8h*)(p);
  v8h hi = *(const v8h*)(p + 16);
  v16h out;
#pragma unroll
  for (int i = 0; i < 8; ++i) { out[i] = lo[i]; out[i + 8] = hi[i]; }
  return out;
}

#define WMMA_F16(a, b, c) \
  __builtin_amdgcn_wmma_f32_16x16x32_f16(false, (a), false, (b), (short)0, (c), false, false)

// ---- Kernel 1: f32 [rows][cols] -> f16 transposed [cols][rows] ----
__global__ void transpose_cvt_kernel(const float* __restrict__ src,
                                     _Float16* __restrict__ dst,
                                     int rows, int cols) {
  int idx = blockIdx.x * 256 + threadIdx.x;
  if (idx >= rows * cols) return;
  int n = idx / rows;
  int k = idx % rows;
  dst[idx] = (_Float16)src[(size_t)k * cols + n];
}

// ---- Kernel 2: qkv = x @ w_qkv + b_qkv, scattered into Q,K (b,h,n,d) and Vt (b,h,d,n) ----
// Fully unrolled k-loop: scheduler hoists loads ahead of WMMAs without rotation movs.
__global__ __launch_bounds__(256) void qkv_gemm_kernel(
    const float* __restrict__ x, const _Float16* __restrict__ wqkvT,
    const float* __restrict__ bqkv,
    _Float16* __restrict__ Q, _Float16* __restrict__ K, _Float16* __restrict__ Vt) {
  const int lane  = threadIdx.x & 31;
  const int strip = blockIdx.x * 8 + (threadIdx.x >> 5);   // 1024*36 = 36864 strips
  const int mt    = strip / 36;
  const int nbase = (strip % 36) * 64;
  const float* abase = x + (size_t)(mt * 16) * H_;
  const _Float16* bbase = wqkvT + (size_t)nbase * H_;

  v8f acc[4] = {};
#pragma unroll
  for (int kk = 0; kk < H_; kk += 32) {
    v16h a = load_a_frag_f32(abase + kk, H_, lane);
    v16h b0 = load_b_frag(bbase + kk, H_, lane);
    v16h b1 = load_b_frag(bbase + (size_t)16 * H_ + kk, H_, lane);
    v16h b2 = load_b_frag(bbase + (size_t)32 * H_ + kk, H_, lane);
    v16h b3 = load_b_frag(bbase + (size_t)48 * H_ + kk, H_, lane);
    acc[0] = WMMA_F16(a, b0, acc[0]);
    acc[1] = WMMA_F16(a, b1, acc[1]);
    acc[2] = WMMA_F16(a, b2, acc[2]);
    acc[3] = WMMA_F16(a, b3, acc[3]);
  }

  const int c = lane & 15, mh = lane >> 4;
#pragma unroll
  for (int t = 0; t < 4; ++t) {
    const int j   = nbase + t * 16 + c;   // column in [0, 2304)
    const int seg = j / H_;               // 0=Q 1=K 2=V (tile never straddles)
    const int jm  = j % H_;
    const int h   = jm / HD_, d = jm % HD_;
    const float bias = bqkv[j];
#pragma unroll
    for (int r = 0; r < 8; ++r) {
      const int m    = mt * 16 + mh * 8 + r;
      const int bi   = m >> 10, nseq = m & 1023;
      const _Float16 hv = (_Float16)(acc[t][r] + bias);
      if (seg == 0)      Q [((size_t)(bi * NH_ + h) * N_ + nseq) * HD_ + d] = hv;
      else if (seg == 1) K [((size_t)(bi * NH_ + h) * N_ + nseq) * HD_ + d] = hv;
      else               Vt[((size_t)(bi * NH_ + h) * HD_ + d) * N_ + nseq] = hv;
    }
  }
}

// ---- Kernel 3: flash-style attention, online softmax, 1 wave per 16-query tile ----
// 8 waves/block share one (b,h) and stream the same K/V chunks (L2/WGP$ reuse).
__global__ __launch_bounds__(256) void attn_kernel(
    const _Float16* __restrict__ Q, const _Float16* __restrict__ K,
    const _Float16* __restrict__ Vt, _Float16* __restrict__ attn) {
  __shared__ __align__(32) _Float16 pstage[8][16 * 32];   // 1 KB per wave

  const int lane = threadIdx.x & 31;
  const int w    = threadIdx.x >> 5;
  const int bh   = blockIdx.x >> 3;             // 0..191
  const int qt   = (blockIdx.x & 7) * 8 + w;    // 0..63
  const _Float16* Qb = Q  + (size_t)bh * N_ * HD_ + (size_t)qt * 16 * HD_;
  const _Float16* Kb = K  + (size_t)bh * N_ * HD_;
  const _Float16* Vb = Vt + (size_t)bh * HD_ * N_;
  _Float16* pl = &pstage[w][0];

  const v16h qa0 = load_a_frag_h(Qb, HD_, lane);       // d = 0..31
  const v16h qa1 = load_a_frag_h(Qb + 32, HD_, lane);  // d = 32..63
  const int c = lane & 15, mh = lane >> 4;

  float mrun[8], lrun[8];
#pragma unroll
  for (int r = 0; r < 8; ++r) { mrun[r] = -3.0e38f; lrun[r] = 0.f; }
  v8f o[4] = {};

  for (int kc = 0; kc < 32; ++kc) {               // 32 chunks of 32 keys
    const _Float16* kchunk = Kb + (size_t)kc * 32 * HD_;
    const _Float16* vchunk = Vb + (size_t)kc * 32;
    __builtin_prefetch(kchunk + 32 * HD_, 0, 1);  // next K chunk -> global_prefetch_b8

    // Issue ALL global loads for this chunk up front: K frags feed the S WMMAs,
    // V frags are consumed ~the whole softmax later (latency fully hidden).
    v16h kb0 = load_b_frag(kchunk, HD_, lane);
    v16h kb1 = load_b_frag(kchunk + 32, HD_, lane);
    v16h kb2 = load_b_frag(kchunk + (size_t)16 * HD_, HD_, lane);
    v16h kb3 = load_b_frag(kchunk + (size_t)16 * HD_ + 32, HD_, lane);
    v16h vb0 = load_b_frag(vchunk, N_, lane);
    v16h vb1 = load_b_frag(vchunk + (size_t)16 * N_, N_, lane);
    v16h vb2 = load_b_frag(vchunk + (size_t)32 * N_, N_, lane);
    v16h vb3 = load_b_frag(vchunk + (size_t)48 * N_, N_, lane);

    // S chunk: 16 q x 32 keys = two C tiles
    v8f s0 = {}, s1 = {};
    s0 = WMMA_F16(qa0, kb0, s0);
    s0 = WMMA_F16(qa1, kb1, s0);
    s1 = WMMA_F16(qa0, kb2, s1);
    s1 = WMMA_F16(qa1, kb3, s1);

    // online softmax update; row m lives across the 16 lanes of one half-wave
#pragma unroll
    for (int r = 0; r < 8; ++r) {
      const float v0 = s0[r] * 0.125f;
      const float v1 = s1[r] * 0.125f;
      float mx = fmaxf(v0, v1);
      mx = fmaxf(mx, __shfl_xor(mx, 1, 32));
      mx = fmaxf(mx, __shfl_xor(mx, 2, 32));
      mx = fmaxf(mx, __shfl_xor(mx, 4, 32));
      mx = fmaxf(mx, __shfl_xor(mx, 8, 32));
      const float nm   = fmaxf(mrun[r], mx);
      const float corr = __expf(mrun[r] - nm);
      mrun[r] = nm;
      const float e0 = __expf(v0 - nm);
      const float e1 = __expf(v1 - nm);
      float rs = e0 + e1;
      rs += __shfl_xor(rs, 1, 32);
      rs += __shfl_xor(rs, 2, 32);
      rs += __shfl_xor(rs, 4, 32);
      rs += __shfl_xor(rs, 8, 32);
      lrun[r] = lrun[r] * corr + rs;
#pragma unroll
      for (int t = 0; t < 4; ++t) o[t][r] *= corr;
      const int m = mh * 8 + r;
      pl[m * 32 + c]      = (_Float16)e0;   // C-layout -> row-major P tile in LDS
      pl[m * 32 + 16 + c] = (_Float16)e1;
    }

    // O += P @ V_chunk  (P read back in A-fragment layout; V frags already resident)
    v16h pa = load_a_frag_h(pl, 32, lane);
    o[0] = WMMA_F16(pa, vb0, o[0]);
    o[1] = WMMA_F16(pa, vb1, o[1]);
    o[2] = WMMA_F16(pa, vb2, o[2]);
    o[3] = WMMA_F16(pa, vb3, o[3]);
  }

  const int bi = bh / NH_, h = bh % NH_;
#pragma unroll
  for (int t = 0; t < 4; ++t) {
    const int d = t * 16 + c;
#pragma unroll
    for (int r = 0; r < 8; ++r) {
      const int nseq = qt * 16 + mh * 8 + r;
      attn[((size_t)(bi * N_ + nseq)) * H_ + h * HD_ + d] = (_Float16)(o[t][r] / lrun[r]);
    }
  }
}

// ---- Kernel 4: out = attn @ w_out + b_out (f32 output), fully unrolled k-loop ----
__global__ __launch_bounds__(256) void out_gemm_kernel(
    const _Float16* __restrict__ attn, const _Float16* __restrict__ woutT,
    const float* __restrict__ bout, float* __restrict__ out) {
  const int lane  = threadIdx.x & 31;
  const int strip = blockIdx.x * 8 + (threadIdx.x >> 5);   // 1024*12 = 12288 strips
  const int mt    = strip / 12;
  const int nbase = (strip % 12) * 64;
  const _Float16* abase = attn + (size_t)(mt * 16) * H_;
  const _Float16* bbase = woutT + (size_t)nbase * H_;

  v8f acc[4] = {};
#pragma unroll
  for (int kk = 0; kk < H_; kk += 32) {
    v16h a = load_a_frag_h(abase + kk, H_, lane);
    v16h b0 = load_b_frag(bbase + kk, H_, lane);
    v16h b1 = load_b_frag(bbase + (size_t)16 * H_ + kk, H_, lane);
    v16h b2 = load_b_frag(bbase + (size_t)32 * H_ + kk, H_, lane);
    v16h b3 = load_b_frag(bbase + (size_t)48 * H_ + kk, H_, lane);
    acc[0] = WMMA_F16(a, b0, acc[0]);
    acc[1] = WMMA_F16(a, b1, acc[1]);
    acc[2] = WMMA_F16(a, b2, acc[2]);
    acc[3] = WMMA_F16(a, b3, acc[3]);
  }

  const int c = lane & 15, mh = lane >> 4;
#pragma unroll
  for (int t = 0; t < 4; ++t) {
    const int j = nbase + t * 16 + c;
    const float bias = bout[j];
#pragma unroll
    for (int r = 0; r < 8; ++r) {
      const int m = mt * 16 + mh * 8 + r;
      out[(size_t)m * H_ + j] = acc[t][r] + bias;
    }
  }
}

extern "C" void kernel_launch(void* const* d_in, const int* in_sizes, int n_in,
                              void* d_out, int out_size, void* d_ws, size_t ws_size,
                              hipStream_t stream) {
  const float* x     = (const float*)d_in[0];
  const float* w_qkv = (const float*)d_in[1];
  const float* b_qkv = (const float*)d_in[2];
  const float* w_out = (const float*)d_in[3];
  const float* b_out = (const float*)d_in[4];
  float* out = (float*)d_out;

  char* ws = (char*)d_ws;
  const size_t SZ_WQKV = (size_t)3 * H_ * H_ * 2;
  const size_t SZ_WOUT = (size_t)H_ * H_ * 2;
  const size_t SZ_QKV  = (size_t)B_ * NH_ * N_ * HD_ * 2;
  size_t off = 0;
  _Float16* wqkvT = (_Float16*)(ws + off); off += SZ_WQKV;
  _Float16* woutT = (_Float16*)(ws + off); off += SZ_WOUT;
  _Float16* Qb    = (_Float16*)(ws + off); off += SZ_QKV;
  _Float16* Kb    = (_Float16*)(ws + off); off += SZ_QKV;
  _Float16* Vt    = (_Float16*)(ws + off); off += SZ_QKV;
  _Float16* attn  = (_Float16*)(ws + off); off += SZ_QKV;

  {
    int n1 = H_ * 3 * H_;
    transpose_cvt_kernel<<<(n1 + 255) / 256, 256, 0, stream>>>(w_qkv, wqkvT, H_, 3 * H_);
    int n2 = H_ * H_;
    transpose_cvt_kernel<<<(n2 + 255) / 256, 256, 0, stream>>>(w_out, woutT, H_, H_);
  }
  qkv_gemm_kernel<<<4608, 256, 0, stream>>>(x, wqkvT, b_qkv, Qb, Kb, Vt);
  attn_kernel<<<B_ * NH_ * 8, 256, 0, stream>>>(Qb, Kb, Vt, attn);
  out_gemm_kernel<<<1536, 256, 0, stream>>>(attn, woutT, b_out, out);
}
